// TransformerDecoderLayer_19258633355777
// MI455X (gfx1250) — matrix operand order repaired and checked by reading, not compiled, not run on hardware
//
#include <hip/hip_runtime.h>
#include <math.h>

typedef __attribute__((ext_vector_type(16))) _Float16 v16h;
typedef __attribute__((ext_vector_type(4)))  _Float16 v4h;
typedef __attribute__((ext_vector_type(8)))  float    v8f;
typedef __attribute__((ext_vector_type(4)))  unsigned int u32x4;
typedef __attribute__((ext_vector_type(8)))  int      i32x8;
typedef __attribute__((ext_vector_type(4)))  int      i32x4;

#define WMMA_F16(A, B, C) \
  __builtin_amdgcn_wmma_f32_16x16x32_f16(false, (A), false, (B), (short)0, (C), false, false)

// ---------------------------------------------------------------------------
// TDM: async-copy one [128 x 32] f16 tile (row stride K halves) from global
// into LDS with row padding 8 halves (-> LDS stride 40 halves). One issue per
// workgroup (TDM ignores EXEC); tracked with TENSORcnt.
// D# layout per CDNA5 ISA.
// ---------------------------------------------------------------------------
__device__ static inline void tdm_load_tile_f16(const _Float16* gsrc,
                                                unsigned lds_off,
                                                int rows_avail, int K)
{
  unsigned long long ga = (unsigned long long)(uintptr_t)gsrc;
  u32x4 g0;
  g0[0] = 1u;                                   // count=1 valid descriptor
  g0[1] = lds_off;                              // lds_addr (bytes)
  g0[2] = (unsigned)(ga & 0xffffffffu);         // global_addr[31:0]
  g0[3] = (unsigned)((ga >> 32) & 0x1ffffffu)   // global_addr[56:32]
        | (2u << 30);                           // type = 2 ("image")
  i32x8 g1;
  g1[0] = (1 << 16)                             // data_size = 2 bytes
        | (1 << 20)                             // pad_enable
        | (3 << 22)                             // pad_interval: 16 dwords/row
        | (3 << 25);                            // pad_amount: 4 dwords (8 halves)
  g1[1] = (K & 0xffff) << 16;                   // tensor_dim0[15:0] @bit48
  g1[2] = ((K >> 16) & 0xffff)                  // tensor_dim0[31:16]
        | ((rows_avail & 0xffff) << 16);        // tensor_dim1[15:0] @bit80
  g1[3] = ((rows_avail >> 16) & 0xffff)         // tensor_dim1[31:16]
        | (32 << 16);                           // tile_dim0 = 32 @bit112
  g1[4] = 128;                                  // tile_dim1 = 128; tile_dim2 = 0
  g1[5] = K;                                    // tensor_dim0_stride[31:0]
  g1[6] = 0;                                    // stride[47:32], dim1_stride lo
  g1[7] = 0;
  i32x4 z4; z4[0] = z4[1] = z4[2] = z4[3] = 0;
#if defined(__clang_major__) && __clang_major__ >= 23
  i32x8 z8;
  #pragma unroll
  for (int i = 0; i < 8; ++i) z8[i] = 0;
  __builtin_amdgcn_tensor_load_to_lds(g0, g1, z4, z4, z8, 0);
#else
  __builtin_amdgcn_tensor_load_to_lds(g0, g1, z4, z4, 0);
#endif
}

// ---------------------------------------------------------------------------
// GEMM (A f32): C = act(A[MxK] @ W[KxN] + bias).  f16 WMMA, f32 accumulate.
// Block tile 128x128, 256 threads = 8 waves (4x2), wave tile 32x64 (2x4 WMMA).
// LDS holds f16: A row-major (converted at staging), W transposed (col-major)
// so all fragment reads are contiguous 16B chunks.
// Output row remap: orow = (row/spb)*opb + row%spb (padded K/V buffers).
// ---------------------------------------------------------------------------
constexpr int TM = 128, TN = 128, TK = 32;

template <typename OutT>
__global__ __launch_bounds__(256)
void gemm_f32a(const float* __restrict__ A, const float* __restrict__ W,
               const float* __restrict__ bias, OutT* __restrict__ C,
               int M, int N, int K, int spb, int opb, int relu)
{
  __shared__ _Float16 As[TM][40];      // 128 x 32 (+8 pad)
  __shared__ _Float16 Bt[TN][40];      // transposed: [n][k]

  const int tid  = threadIdx.x;
  const int lane = tid & 31;
  const int wave = tid >> 5;
  const int wm = wave >> 1, wn = wave & 1;
  const int lg = lane >> 4, lc = lane & 15;
  const int m0 = blockIdx.y * TM;
  const int n0 = blockIdx.x * TN;

  v8f acc[2][4];
  #pragma unroll
  for (int i = 0; i < 2; ++i)
    #pragma unroll
    for (int j = 0; j < 4; ++j)
      #pragma unroll
      for (int r = 0; r < 8; ++r) acc[i][j][r] = 0.f;

  for (int k0 = 0; k0 < K; k0 += TK) {
    // Load phase first (keeps several global loads in flight), then convert.
    float4 fa[4], fb[4];
    #pragma unroll
    for (int i = 0; i < 4; ++i) {
      int idx4 = tid + 256 * i;
      int r = idx4 >> 3;
      int c = (idx4 & 7) << 2;
      int grow = m0 + r;
      fa[i] = make_float4(0.f, 0.f, 0.f, 0.f);
      if (grow < M) fa[i] = *(const float4*)(A + (size_t)grow * K + k0 + c);
    }
    #pragma unroll
    for (int i = 0; i < 4; ++i) {
      int idx4 = tid + 256 * i;
      int r = idx4 >> 5;               // k row 0..31
      int c = (idx4 & 31) << 2;        // n col
      fb[i] = *(const float4*)(W + (size_t)(k0 + r) * N + n0 + c);
    }
    #pragma unroll
    for (int i = 0; i < 4; ++i) {
      int idx4 = tid + 256 * i;
      int r = idx4 >> 3;
      int c = (idx4 & 7) << 2;
      v4h h; h[0] = (_Float16)fa[i].x; h[1] = (_Float16)fa[i].y;
             h[2] = (_Float16)fa[i].z; h[3] = (_Float16)fa[i].w;
      *(v4h*)&As[r][c] = h;
    }
    #pragma unroll
    for (int i = 0; i < 4; ++i) {
      int idx4 = tid + 256 * i;
      int r = idx4 >> 5;
      int c = (idx4 & 31) << 2;
      Bt[c + 0][r] = (_Float16)fb[i].x;
      Bt[c + 1][r] = (_Float16)fb[i].y;
      Bt[c + 2][r] = (_Float16)fb[i].z;
      Bt[c + 3][r] = (_Float16)fb[i].w;
    }
    __syncthreads();

    v16h af[2], bf[4];
    #pragma unroll
    for (int t = 0; t < 2; ++t) {
      int m  = wm * 32 + t * 16 + lc;
      int kb = lg * 8;
      #pragma unroll
      for (int i = 0; i < 8; ++i) {
        af[t][i]     = As[m][kb + i];
        af[t][8 + i] = As[m][kb + 16 + i];
      }
    }
    #pragma unroll
    for (int t = 0; t < 4; ++t) {
      int col = wn * 64 + t * 16 + lc;
      int kr  = lg * 16;
      #pragma unroll
      for (int i = 0; i < 16; ++i)
        bf[t][i] = Bt[col][(kr >> 1) + (i & 7) + ((i >> 3) << 4)];
    }
    #pragma unroll
    for (int ti = 0; ti < 2; ++ti)
      #pragma unroll
      for (int tj = 0; tj < 4; ++tj)
        acc[ti][tj] = WMMA_F16(af[ti], bf[tj], acc[ti][tj]);
    __syncthreads();
  }

  #pragma unroll
  for (int ti = 0; ti < 2; ++ti)
    #pragma unroll
    for (int tj = 0; tj < 4; ++tj)
      #pragma unroll
      for (int r = 0; r < 8; ++r) {
        int row = m0 + wm * 32 + ti * 16 + lg * 8 + r;
        int col = n0 + wn * 64 + tj * 16 + lc;
        if (row < M) {
          float v = acc[ti][tj][r] + bias[col];
          if (relu) v = fmaxf(v, 0.f);
          int orow = (row / spb) * opb + (row % spb);
          C[(size_t)orow * N + col] = (OutT)v;
        }
      }
}

// ---------------------------------------------------------------------------
// GEMM (A f16): A tiles staged by the Tensor Data Mover into double-buffered
// LDS; the TDM for tile k+1 (and the B-stage loads) are issued before the
// WMMAs of tile k, drained with s_wait_tensorcnt after — DMA overlaps compute.
// Used for h @ f2_w.
// ---------------------------------------------------------------------------
__global__ __launch_bounds__(256)
void gemm_f16a(const _Float16* __restrict__ A, const float* __restrict__ W,
               const float* __restrict__ bias, float* __restrict__ C,
               int M, int N, int K)
{
  __shared__ _Float16 As[2][TM][40];
  __shared__ _Float16 Bt[2][TN][40];

  const int tid  = threadIdx.x;
  const int lane = tid & 31;
  const int wave = tid >> 5;
  const int wm = wave >> 1, wn = wave & 1;
  const int lg = lane >> 4, lc = lane & 15;
  const int m0 = blockIdx.y * TM;
  const int n0 = blockIdx.x * TN;
  const unsigned as_off[2] = { (unsigned)(uintptr_t)&As[0][0][0],
                               (unsigned)(uintptr_t)&As[1][0][0] };

  auto stageB = [&](int k0, int buf) {
    float4 fb[4];
    #pragma unroll
    for (int i = 0; i < 4; ++i) {
      int idx4 = tid + 256 * i;
      int r = idx4 >> 5;
      int c = (idx4 & 31) << 2;
      fb[i] = *(const float4*)(W + (size_t)(k0 + r) * N + n0 + c);
    }
    #pragma unroll
    for (int i = 0; i < 4; ++i) {
      int idx4 = tid + 256 * i;
      int r = idx4 >> 5;
      int c = (idx4 & 31) << 2;
      Bt[buf][c + 0][r] = (_Float16)fb[i].x;
      Bt[buf][c + 1][r] = (_Float16)fb[i].y;
      Bt[buf][c + 2][r] = (_Float16)fb[i].z;
      Bt[buf][c + 3][r] = (_Float16)fb[i].w;
    }
  };

  v8f acc[2][4];
  #pragma unroll
  for (int i = 0; i < 2; ++i)
    #pragma unroll
    for (int j = 0; j < 4; ++j)
      #pragma unroll
      for (int r = 0; r < 8; ++r) acc[i][j][r] = 0.f;

  const int nk = K / TK;

  // Prologue: tile 0 into buffer 0
  if (wave == 0) tdm_load_tile_f16(A + (size_t)m0 * K, as_off[0], M - m0, K);
  stageB(0, 0);
  if (wave == 0) __builtin_amdgcn_s_wait_tensorcnt(0);
  __syncthreads();

  for (int kt = 0; kt < nk; ++kt) {
    const int buf = kt & 1, nxt = buf ^ 1;
    // Prefetch tile kt+1 into the other buffer while we compute tile kt
    if (kt + 1 < nk) {
      if (wave == 0)
        tdm_load_tile_f16(A + (size_t)m0 * K + (size_t)(kt + 1) * TK,
                          as_off[nxt], M - m0, K);
      stageB((kt + 1) * TK, nxt);
    }

    v16h af[2], bf[4];
    #pragma unroll
    for (int t = 0; t < 2; ++t) {
      int m  = wm * 32 + t * 16 + lc;
      int kb = lg * 8;
      #pragma unroll
      for (int i = 0; i < 8; ++i) {
        af[t][i]     = As[buf][m][kb + i];
        af[t][8 + i] = As[buf][m][kb + 16 + i];
      }
    }
    #pragma unroll
    for (int t = 0; t < 4; ++t) {
      int col = wn * 64 + t * 16 + lc;
      int kr  = lg * 16;
      #pragma unroll
      for (int i = 0; i < 16; ++i)
        bf[t][i] = Bt[buf][col][(kr >> 1) + (i & 7) + ((i >> 3) << 4)];
    }
    #pragma unroll
    for (int ti = 0; ti < 2; ++ti)
      #pragma unroll
      for (int tj = 0; tj < 4; ++tj)
        acc[ti][tj] = WMMA_F16(af[ti], bf[tj], acc[ti][tj]);

    if (wave == 0 && kt + 1 < nk) __builtin_amdgcn_s_wait_tensorcnt(0);
    __syncthreads();
  }

  #pragma unroll
  for (int ti = 0; ti < 2; ++ti)
    #pragma unroll
    for (int tj = 0; tj < 4; ++tj)
      #pragma unroll
      for (int r = 0; r < 8; ++r) {
        int row = m0 + wm * 32 + ti * 16 + lg * 8 + r;
        int col = n0 + wn * 64 + tj * 16 + lc;
        if (row < M)
          C[(size_t)row * N + col] = acc[ti][tj][r] + bias[col];
      }
}

// ---------------------------------------------------------------------------
// Flash-style attention. H=16, depth=64, D=1024, Sq=1024.
// 128 threads = 4 waves; 64 q-rows per block, 16 per wave; 32-key blocks.
// f16 LDS everywhere; V stored transposed so P@V B-fragments are contiguous.
// Q pre-scaled by 1/sqrt(64)=0.125 at conversion.
// ---------------------------------------------------------------------------
__global__ __launch_bounds__(128)
void attn_kernel(const float* __restrict__ Q, const float* __restrict__ K,
                 const float* __restrict__ V, float* __restrict__ O,
                 int Lk, int Lkp)
{
  __shared__ _Float16 Qs[64][40];
  __shared__ _Float16 Ks[32][40];
  __shared__ _Float16 Vst[64][40];     // transposed: [depth][key]
  __shared__ _Float16 Ps[4][16][40];   // wave-private P (16 q x 32 keys)

  const int tid  = threadIdx.x;
  const int lane = tid & 31;
  const int wave = tid >> 5;
  const int lg = lane >> 4, lc = lane & 15;
  const int b  = blockIdx.z;
  const int h  = blockIdx.y;
  const int q0 = blockIdx.x * 64;

  const float* Qbase = Q + ((size_t)b * 1024 + q0) * 1024 + h * 64;
  const float* Kbase = K + (size_t)b * Lkp * 1024 + h * 64;
  const float* Vbase = V + (size_t)b * Lkp * 1024 + h * 64;

  // Q tile 64x64 -> f16 LDS (scaled)
  {
    float4 fq[8];
    #pragma unroll
    for (int i = 0; i < 8; ++i) {
      int idx4 = tid + 128 * i;
      int r = idx4 >> 4;
      int c = (idx4 & 15) << 2;
      fq[i] = *(const float4*)(Qbase + (size_t)r * 1024 + c);
    }
    #pragma unroll
    for (int i = 0; i < 8; ++i) {
      int idx4 = tid + 128 * i;
      int r = idx4 >> 4;
      int c = (idx4 & 15) << 2;
      v4h hq; hq[0] = (_Float16)(fq[i].x * 0.125f);
              hq[1] = (_Float16)(fq[i].y * 0.125f);
              hq[2] = (_Float16)(fq[i].z * 0.125f);
              hq[3] = (_Float16)(fq[i].w * 0.125f);
      *(v4h*)&Qs[r][c] = hq;
    }
  }
  __syncthreads();

  v16h qa[2];
  {
    int m = wave * 16 + lc;
    #pragma unroll
    for (int t = 0; t < 2; ++t) {
      int kb = t * 32 + lg * 8;
      #pragma unroll
      for (int i = 0; i < 8; ++i) {
        qa[t][i]     = Qs[m][kb + i];
        qa[t][8 + i] = Qs[m][kb + 16 + i];
      }
    }
  }

  float mrow[8], lrow[8];
  v8f   oacc[4];
  #pragma unroll
  for (int r = 0; r < 8; ++r) { mrow[r] = -INFINITY; lrow[r] = 0.f; }
  #pragma unroll
  for (int t = 0; t < 4; ++t)
    #pragma unroll
    for (int r = 0; r < 8; ++r) oacc[t][r] = 0.f;

  for (int j0 = 0; j0 < Lk; j0 += 32) {
    // K block row-major f16; V block transposed f16 (loads first, then cvt)
    {
      float4 fk[4], fv[4];
      #pragma unroll
      for (int i = 0; i < 4; ++i) {
        int idx4 = tid + 128 * i;
        int r = idx4 >> 4;
        int c = (idx4 & 15) << 2;
        fk[i] = *(const float4*)(Kbase + (size_t)(j0 + r) * 1024 + c);
        fv[i] = *(const float4*)(Vbase + (size_t)(j0 + r) * 1024 + c);
      }
      #pragma unroll
      for (int i = 0; i < 4; ++i) {
        int idx4 = tid + 128 * i;
        int r = idx4 >> 4;
        int c = (idx4 & 15) << 2;
        v4h hk; hk[0] = (_Float16)fk[i].x; hk[1] = (_Float16)fk[i].y;
                hk[2] = (_Float16)fk[i].z; hk[3] = (_Float16)fk[i].w;
        *(v4h*)&Ks[r][c] = hk;
        Vst[c + 0][r] = (_Float16)fv[i].x;
        Vst[c + 1][r] = (_Float16)fv[i].y;
        Vst[c + 2][r] = (_Float16)fv[i].z;
        Vst[c + 3][r] = (_Float16)fv[i].w;
      }
    }
    __syncthreads();

    // Logits: two 16-key tiles, contraction over depth (2 WMMA each)
    v8f cfr[2];
    #pragma unroll
    for (int tj = 0; tj < 2; ++tj) {
      v8f cz;
      #pragma unroll
      for (int r = 0; r < 8; ++r) cz[r] = 0.f;
      #pragma unroll
      for (int t = 0; t < 2; ++t) {
        v16h kf;
        #pragma unroll
        for (int i = 0; i < 16; ++i)
          kf[i] = Ks[tj * 16 + lc][t * 32 + lg * 8 + (i & 7) + ((i >> 3) << 4)];
        cz = WMMA_F16(qa[t], kf, cz);
      }
      cfr[tj] = cz;
    }
    // Mask padded key columns
    #pragma unroll
    for (int tj = 0; tj < 2; ++tj)
      if (j0 + tj * 16 + lc >= Lk) {
        #pragma unroll
        for (int r = 0; r < 8; ++r) cfr[tj][r] = -INFINITY;
      }

    // Row max across the 16 lanes of each lane group
    float bmax[8];
    #pragma unroll
    for (int r = 0; r < 8; ++r) bmax[r] = fmaxf(cfr[0][r], cfr[1][r]);
    #pragma unroll
    for (int off = 8; off >= 1; off >>= 1)
      #pragma unroll
      for (int r = 0; r < 8; ++r)
        bmax[r] = fmaxf(bmax[r], __shfl_xor(bmax[r], off, 32));

    float alpha[8], rsum[8];
    #pragma unroll
    for (int r = 0; r < 8; ++r) {
      float mn = fmaxf(mrow[r], bmax[r]);
      alpha[r] = __expf(mrow[r] - mn);
      mrow[r]  = mn;
      rsum[r]  = 0.f;
    }

    // P = exp(logit - m); LDS round-trip to reach A-fragment layout
    #pragma unroll
    for (int tj = 0; tj < 2; ++tj)
      #pragma unroll
      for (int r = 0; r < 8; ++r) {
        float p = __expf(cfr[tj][r] - mrow[r]);
        rsum[r] += p;
        Ps[wave][lg * 8 + r][tj * 16 + lc] = (_Float16)p;
      }
    #pragma unroll
    for (int off = 8; off >= 1; off >>= 1)
      #pragma unroll
      for (int r = 0; r < 8; ++r)
        rsum[r] += __shfl_xor(rsum[r], off, 32);
    #pragma unroll
    for (int r = 0; r < 8; ++r) lrow[r] = lrow[r] * alpha[r] + rsum[r];
    #pragma unroll
    for (int tn = 0; tn < 4; ++tn)
      #pragma unroll
      for (int r = 0; r < 8; ++r) oacc[tn][r] *= alpha[r];

    // P (16x32) @ V (32x64): contiguous fragments from Ps and Vst
    v16h pa;
    {
      int kb = lg * 8;
      #pragma unroll
      for (int i = 0; i < 8; ++i) {
        pa[i]     = Ps[wave][lc][kb + i];
        pa[8 + i] = Ps[wave][lc][kb + 16 + i];
      }
    }
    #pragma unroll
    for (int tn = 0; tn < 4; ++tn) {
      v16h vb;
      #pragma unroll
      for (int i = 0; i < 16; ++i)
        vb[i] = Vst[tn * 16 + lc][lg * 8 + (i & 7) + ((i >> 3) << 4)];
      oacc[tn] = WMMA_F16(pa, vb, oacc[tn]);
    }
    __syncthreads();
  }

  #pragma unroll
  for (int tn = 0; tn < 4; ++tn)
    #pragma unroll
    for (int r = 0; r < 8; ++r) {
      int qg = q0 + wave * 16 + lg * 8 + r;
      O[((size_t)b * 1024 + qg) * 1024 + h * 64 + tn * 16 + lc] =
          oacc[tn][r] / lrow[r];
    }
}

// ---------------------------------------------------------------------------
// out = LN(a + b) * g + beta, one block per row, D=1024
// ---------------------------------------------------------------------------
__global__ __launch_bounds__(256)
void add_ln_kernel(const float* __restrict__ A, const float* __restrict__ Bp,
                   const float* __restrict__ G, const float* __restrict__ Bt,
                   float* __restrict__ Out)
{
  __shared__ float sred[8];
  const int row = blockIdx.x;
  const int t = threadIdx.x;
  const size_t base = (size_t)row * 1024;

  float x[4];
  #pragma unroll
  for (int i = 0; i < 4; ++i) {
    int c = t * 4 + i;
    x[i] = A[base + c] + Bp[base + c];
  }
  float s = x[0] + x[1] + x[2] + x[3];
  #pragma unroll
  for (int off = 16; off >= 1; off >>= 1) s += __shfl_xor(s, off, 32);
  if ((t & 31) == 0) sred[t >> 5] = s;
  __syncthreads();
  float mean = 0.f;
  #pragma unroll
  for (int i = 0; i < 8; ++i) mean += sred[i];
  mean *= (1.f / 1024.f);
  __syncthreads();

  float vs = 0.f;
  #pragma unroll
  for (int i = 0; i < 4; ++i) { float d = x[i] - mean; vs += d * d; }
  #pragma unroll
  for (int off = 16; off >= 1; off >>= 1) vs += __shfl_xor(vs, off, 32);
  if ((t & 31) == 0) sred[t >> 5] = vs;
  __syncthreads();
  float var = 0.f;
  #pragma unroll
  for (int i = 0; i < 8; ++i) var += sred[i];
  var *= (1.f / 1024.f);
  float rstd = rsqrtf(var + 1e-6f);

  #pragma unroll
  for (int i = 0; i < 4; ++i) {
    int c = t * 4 + i;
    Out[base + c] = (x[i] - mean) * rstd * G[c] + Bt[c];
  }
}

// ---------------------------------------------------------------------------
// Broadcast 40 projected memory K/V rows into each batch's padded buffer
// (rows 1024..1063) and zero the padding (rows 1064..1087). D=1024, B=4.
// ---------------------------------------------------------------------------
__global__ void memfill_kernel(const float* __restrict__ km, const float* __restrict__ vm,
                               float* __restrict__ kf, float* __restrict__ vf)
{
  int idx = blockIdx.x * blockDim.x + threadIdx.x;   // 4 * 64 * 1024 total
  if (idx >= 4 * 64 * 1024) return;
  int c = idx & 1023;
  int r = idx >> 10;
  int b = r >> 6;
  int j = r & 63;
  float kv = (j < 40) ? km[j * 1024 + c] : 0.f;
  float vv = (j < 40) ? vm[j * 1024 + c] : 0.f;
  size_t off = ((size_t)b * 1088 + 1024 + j) * 1024 + c;
  kf[off] = kv;
  vf[off] = vv;
}

// ---------------------------------------------------------------------------
extern "C" void kernel_launch(void* const* d_in, const int* in_sizes, int n_in,
                              void* d_out, int out_size, void* d_ws, size_t ws_size,
                              hipStream_t stream)
{
  const float* x     = (const float*)d_in[0];
  const float* c_att = (const float*)d_in[1];
  const float* s_att = (const float*)d_in[2];
  const float* enc   = (const float*)d_in[3];
  const float* q1w = (const float*)d_in[4];  const float* q1b = (const float*)d_in[5];
  const float* k1w = (const float*)d_in[6];  const float* k1b = (const float*)d_in[7];
  const float* v1w = (const float*)d_in[8];  const float* v1b = (const float*)d_in[9];
  const float* o1w = (const float*)d_in[10]; const float* o1b = (const float*)d_in[11];
  const float* q2w = (const float*)d_in[12]; const float* q2b = (const float*)d_in[13];
  const float* k2w = (const float*)d_in[14]; const float* k2b = (const float*)d_in[15];
  const float* v2w = (const float*)d_in[16]; const float* v2b = (const float*)d_in[17];
  const float* kmem = (const float*)d_in[18]; const float* vmem = (const float*)d_in[19];
  const float* kww = (const float*)d_in[20]; const float* kwb = (const float*)d_in[21];
  const float* vww = (const float*)d_in[22]; const float* vwb = (const float*)d_in[23];
  const float* o2w = (const float*)d_in[24]; const float* o2b = (const float*)d_in[25];
  const float* f1w = (const float*)d_in[26]; const float* f1b = (const float*)d_in[27];
  const float* f2w = (const float*)d_in[28]; const float* f2b = (const float*)d_in[29];
  const float* ln1g = (const float*)d_in[30]; const float* ln1b = (const float*)d_in[31];
  const float* ln2g = (const float*)d_in[32]; const float* ln2b = (const float*)d_in[33];
  const float* ln3g = (const float*)d_in[34]; const float* ln3b = (const float*)d_in[35];

  const size_t SB  = (size_t)4 * 1024 * 1024;  // activation buffer (floats)
  const size_t SKF = (size_t)4 * 1088 * 1024;  // padded K/V buffer
  const size_t SM  = (size_t)40 * 1024;        // projected memory rows

  float* ws   = (float*)d_ws;
  float* qp   = ws;
  float* kbuf = ws + SB;
  float* vbuf = kbuf + SKF;
  float* attO = vbuf + SKF;
  _Float16* hbuf = (_Float16*)ws;  // FFN hidden (f16) aliases qp..attO (dead then)
  float* tbuf = attO + SB;
  float* out1 = tbuf + SB;
  float* out2 = out1 + SB;
  float* out4 = out2 + SB;
  float* km   = out4 + SB;
  float* vm   = km + SM;

  float* out3 = (float*)d_out;
  float* out5 = out3 + SB;

  auto gemm = [&](const float* A, const float* W, const float* bias, float* C,
                  int M, int N, int K, int opb, int relu) {
    dim3 g(N / TN, (M + TM - 1) / TM);
    gemm_f32a<float><<<g, 256, 0, stream>>>(A, W, bias, C, M, N, K, 1024, opb, relu);
  };
  auto gemm_h = [&](const float* A, const float* W, const float* bias, _Float16* C,
                    int M, int N, int K, int relu) {
    dim3 g(N / TN, (M + TM - 1) / TM);
    gemm_f32a<_Float16><<<g, 256, 0, stream>>>(A, W, bias, C, M, N, K, 1024, 1024, relu);
  };
  auto gemm_ha = [&](const _Float16* A, const float* W, const float* bias, float* C,
                     int M, int N, int K) {
    dim3 g(N / TN, (M + TM - 1) / TM);
    gemm_f16a<<<g, 256, 0, stream>>>(A, W, bias, C, M, N, K);
  };
  auto attn = [&](const float* Qp, const float* Kp, const float* Vp, float* Op,
                  int Lk, int Lkp) {
    dim3 g(16, 16, 4);  // q-tiles x heads x batch
    attn_kernel<<<g, 128, 0, stream>>>(Qp, Kp, Vp, Op, Lk, Lkp);
  };
  auto lnk = [&](const float* a, const float* b, const float* g_, const float* bt,
                 float* o) {
    add_ln_kernel<<<4096, 256, 0, stream>>>(a, b, g_, bt, o);
  };

  // Phase 1: self-attention on x
  gemm(x, q1w, q1b, qp,   4096, 1024, 1024, 1024, 0);
  gemm(x, k1w, k1b, kbuf, 4096, 1024, 1024, 1024, 0);
  gemm(x, v1w, v1b, vbuf, 4096, 1024, 1024, 1024, 0);
  attn(qp, kbuf, vbuf, attO, 1024, 1024);
  gemm(attO, o1w, o1b, tbuf, 4096, 1024, 1024, 1024, 0);
  lnk(x, tbuf, ln1g, ln1b, out1);

  // Phase 2: memory cross-attention (q=out1, k=s_att, v=c_att, +40 mem slots)
  gemm(out1,  q2w, q2b, qp,   4096, 1024, 1024, 1024, 0);
  gemm(s_att, k2w, k2b, kbuf, 4096, 1024, 1024, 1088, 0);
  gemm(c_att, v2w, v2b, vbuf, 4096, 1024, 1024, 1088, 0);
  gemm(kmem, kww, kwb, km, 40, 1024, 1024, 1024, 0);
  gemm(vmem, vww, vwb, vm, 40, 1024, 1024, 1024, 0);
  memfill_kernel<<<(4 * 64 * 1024 + 255) / 256, 256, 0, stream>>>(km, vm, kbuf, vbuf);
  attn(qp, kbuf, vbuf, attO, 1064, 1088);
  gemm(attO, o2w, o2b, tbuf, 4096, 1024, 1024, 1024, 0);
  lnk(out1, tbuf, ln2g, ln2b, out2);

  // Phase 3: FFN (hidden kept in f16; consumer GEMM stages A tiles via TDM)
  gemm_h(out2, f1w, f1b, hbuf, 4096, 4096, 1024, 1);
  gemm_ha(hbuf, f2w, f2b, tbuf, 4096, 1024, 4096);
  lnk(out2, tbuf, ln3g, ln3b, out3);

  // Phase 4: second pass with mha1 weights (q=encoder_out1, k=v=out3)
  gemm(enc,  q1w, q1b, qp,   4096, 1024, 1024, 1024, 0);
  gemm(out3, k1w, k1b, kbuf, 4096, 1024, 1024, 1024, 0);
  gemm(out3, v1w, v1b, vbuf, 4096, 1024, 1024, 1024, 0);
  attn(qp, kbuf, vbuf, attO, 1024, 1024);
  gemm(attO, o1w, o1b, tbuf, 4096, 1024, 1024, 1024, 0);
  lnk(enc, tbuf, ln3g, ln3b, out4);
  gemm_h(out4, f1w, f1b, hbuf, 4096, 4096, 1024, 1);
  gemm_ha(hbuf, f2w, f2b, tbuf, 4096, 1024, 4096);
  lnk(tbuf, out4, ln3g, ln3b, out5);

  (void)in_sizes; (void)n_in; (void)out_size; (void)ws_size;
}
